// SNN_Model_39599598469693
// MI455X (gfx1250) — matrix operand order repaired and checked by reading, not compiled, not run on hardware
//
#include <hip/hip_runtime.h>
#include <hip/hip_bf16.h>

// SNN (3-layer LIF) fully fused for gfx1250.
//   Pass 1 (tiny): pack w1/w2/w3 into bf16 WMMA B-fragment layout in d_ws.
//   Pass 2: cur1 = x@w1^T + b1 via bf16 WMMA, then 10 recurrent steps with all
//   membrane state in registers; spikes relayed through wave-private LDS
//   (C-layout -> bf16 A-layout) and layers 2/3 also via bf16 WMMA.

#define IN_DIM 128
#define H1D    64
#define H2D    32
#define NC     4
#define STEPS  10

typedef __attribute__((ext_vector_type(16))) __bf16 v16bf;
typedef __attribute__((ext_vector_type(8)))  float  v8f;

union V16 {
    v16bf v;
    __bf16 bf[16];
    unsigned short u[16];
    uint4 q[2];
};

__device__ __forceinline__ unsigned short f2bf(float f) {
    unsigned b = __builtin_bit_cast(unsigned, f);
    return (unsigned short)((b + 0x7FFFu + ((b >> 16) & 1u)) >> 16);
}

// ---------------------------------------------------------------------------
// Fragment packing. 21 fragments total, each 32 lanes x 16 bf16 (32 B/lane):
//   frag 0..15  : w1  (ks = frag>>2, t = frag&3)   [64 x 128]
//   frag 16..19 : w2  (ks = (frag-16)>>1, t = (frag-16)&1)  [32 x 64]
//   frag 20     : w3  zero-padded to 16 output columns      [4 x 32]
// B layout (bf16, KxN=Kx16): lane holds column n = lane&15; elems 0..7 carry
// K = kb+hi*8+0..7, elems 8..15 carry K = kb+16+hi*8+0..7.
// ---------------------------------------------------------------------------
__global__ void pack_weights(const float* __restrict__ w1,
                             const float* __restrict__ w2,
                             const float* __restrict__ w3,
                             unsigned short* __restrict__ ws) {
    const int tid  = threadIdx.x;          // 0..671
    const int lane = tid & 31;
    const int frag = tid >> 5;
    const int nl   = lane & 15;
    const int hi   = lane >> 4;
    unsigned short* dst = ws + (size_t)tid * 16;

    if (frag < 16) {
        const int ks = frag >> 2, t = frag & 3;
        const float* wr = w1 + (size_t)(t * 16 + nl) * IN_DIM + ks * 32 + hi * 8;
        #pragma unroll
        for (int e = 0; e < 16; ++e) dst[e] = f2bf(wr[e < 8 ? e : e + 8]);
    } else if (frag < 20) {
        const int f = frag - 16, ks = f >> 1, t = f & 1;
        const float* wr = w2 + (size_t)(t * 16 + nl) * H1D + ks * 32 + hi * 8;
        #pragma unroll
        for (int e = 0; e < 16; ++e) dst[e] = f2bf(wr[e < 8 ? e : e + 8]);
    } else {
        if (nl < NC) {
            const float* wr = w3 + (size_t)nl * H2D + hi * 8;
            #pragma unroll
            for (int e = 0; e < 16; ++e) dst[e] = f2bf(wr[e < 8 ? e : e + 8]);
        } else {
            #pragma unroll
            for (int e = 0; e < 16; ++e) dst[e] = 0;
        }
    }
}

// ---------------------------------------------------------------------------
// Main fused kernel: 8 waves/block, 16 batch rows per wave.
// ---------------------------------------------------------------------------
__launch_bounds__(256)
__global__ void snn_fused(const float* __restrict__ x,
                          const float* __restrict__ b1,
                          const float* __restrict__ b2,
                          const float* __restrict__ b3,
                          const v16bf* __restrict__ wfrag,
                          float* __restrict__ out) {
    __shared__ __align__(16) unsigned short s_spk1[8][16][H1D];
    __shared__ __align__(16) unsigned short s_spk2[8][16][H2D];

    const int lane = threadIdx.x & 31;
    const int wv   = threadIdx.x >> 5;
    const int nl   = lane & 15;
    const int hi   = lane >> 4;
    const int row0 = blockIdx.x * 128 + wv * 16;

    unsigned short (*spk1)[H1D] = s_spk1[wv];
    unsigned short (*spk2)[H2D] = s_spk2[wv];

    // persistent B fragments for the time loop (pre-packed, direct b128 loads)
    v16bf bw2[2][2];
    #pragma unroll
    for (int ks = 0; ks < 2; ++ks)
        #pragma unroll
        for (int t = 0; t < 2; ++t)
            bw2[ks][t] = wfrag[(16 + ks * 2 + t) * 32 + lane];
    const v16bf bw3 = wfrag[20 * 32 + lane];

    // ---------- fc1: cur1 = x @ w1^T + b1  ([16 x 64] = 4 accum tiles) ----------
    v8f cur1[4];
    #pragma unroll
    for (int t = 0; t < 4; ++t)
        #pragma unroll
        for (int r = 0; r < 8; ++r) cur1[t][r] = 0.0f;

    const float* xrow = x + (size_t)(row0 + nl) * IN_DIM;   // A row m = nl
    #pragma unroll
    for (int ks = 0; ks < 4; ++ks) {
        const int kb = ks * 32;
        // bf16 A (16x32): elems 0..7 -> K = kb+hi*8+0..7, 8..15 -> +16..23
        V16 a;
        const float* xr = xrow + kb + hi * 8;
        #pragma unroll
        for (int e = 0; e < 8; ++e) a.bf[e]     = (__bf16)xr[e];
        #pragma unroll
        for (int e = 0; e < 8; ++e) a.bf[8 + e] = (__bf16)xr[16 + e];
        #pragma unroll
        for (int t = 0; t < 4; ++t) {
            const v16bf bw1 = wfrag[(ks * 4 + t) * 32 + lane];
            cur1[t] = __builtin_amdgcn_wmma_f32_16x16x32_bf16(
                false, a.v, false, bw1, (short)0, cur1[t], false, false);
        }
    }
    #pragma unroll
    for (int t = 0; t < 4; ++t) {
        const float bb = b1[t * 16 + nl];
        #pragma unroll
        for (int r = 0; r < 8; ++r) cur1[t][r] += bb;
    }

    // bias splats for layers 2 / 3
    v8f b2v[2], b3v;
    #pragma unroll
    for (int t = 0; t < 2; ++t) {
        const float bb = b2[t * 16 + nl];
        #pragma unroll
        for (int r = 0; r < 8; ++r) b2v[t][r] = bb;
    }
    {
        const float bb = (nl < NC) ? b3[nl] : 0.0f;
        #pragma unroll
        for (int r = 0; r < 8; ++r) b3v[r] = bb;
    }

    // ---------- recurrent state ----------
    v8f mem1[4], mem2[2], mem3, ssum;
    #pragma unroll
    for (int t = 0; t < 4; ++t)
        #pragma unroll
        for (int r = 0; r < 8; ++r) mem1[t][r] = 0.0f;
    #pragma unroll
    for (int t = 0; t < 2; ++t)
        #pragma unroll
        for (int r = 0; r < 8; ++r) mem2[t][r] = 0.0f;
    #pragma unroll
    for (int r = 0; r < 8; ++r) { mem3[r] = 0.0f; ssum[r] = 0.0f; }

    // ---------- 10 LIF timesteps ----------
    for (int step = 0; step < STEPS; ++step) {
        // layer-1 LIF (reset-by-subtraction) + spike scatter to LDS as bf16
        #pragma unroll
        for (int t = 0; t < 4; ++t) {
            #pragma unroll
            for (int r = 0; r < 8; ++r) {
                float mv  = mem1[t][r];
                float rst = (mv > 1.0f) ? 1.0f : 0.0f;
                mv = 0.9f * mv + cur1[t][r] - rst;
                mem1[t][r] = mv;
                spk1[r + hi * 8][t * 16 + nl] =
                    (mv > 1.0f) ? (unsigned short)0x3F80 : (unsigned short)0;
            }
        }
        // wave-private slice; LDS in-order per wave, just drain DScnt
        asm volatile("s_wait_dscnt 0" ::: "memory");

        // layer 2: spk1[16x64] @ w2^T -> [16x32]
        v8f acc2[2] = { b2v[0], b2v[1] };
        #pragma unroll
        for (int ks = 0; ks < 2; ++ks) {
            V16 a;
            const unsigned short* ar = &spk1[nl][ks * 32 + hi * 8];
            a.q[0] = *reinterpret_cast<const uint4*>(ar);        // K +0..7
            a.q[1] = *reinterpret_cast<const uint4*>(ar + 16);   // K +16..23
            #pragma unroll
            for (int t = 0; t < 2; ++t)
                acc2[t] = __builtin_amdgcn_wmma_f32_16x16x32_bf16(
                    false, a.v, false, bw2[ks][t], (short)0, acc2[t], false, false);
        }

        // layer-2 LIF + spike scatter
        #pragma unroll
        for (int t = 0; t < 2; ++t) {
            #pragma unroll
            for (int r = 0; r < 8; ++r) {
                float mv  = mem2[t][r];
                float rst = (mv > 1.0f) ? 1.0f : 0.0f;
                mv = 0.9f * mv + acc2[t][r] - rst;
                mem2[t][r] = mv;
                spk2[r + hi * 8][t * 16 + nl] =
                    (mv > 1.0f) ? (unsigned short)0x3F80 : (unsigned short)0;
            }
        }
        asm volatile("s_wait_dscnt 0" ::: "memory");

        // layer 3: spk2[16x32] @ w3^T(padded) -> [16x16], C preloaded with b3
        V16 a3;
        const unsigned short* ar3 = &spk2[nl][hi * 8];
        a3.q[0] = *reinterpret_cast<const uint4*>(ar3);
        a3.q[1] = *reinterpret_cast<const uint4*>(ar3 + 16);
        v8f cur3 = __builtin_amdgcn_wmma_f32_16x16x32_bf16(
            false, a3.v, false, bw3, (short)0, b3v, false, false);

        // leaky integrator (no spike) + running sum
        #pragma unroll
        for (int r = 0; r < 8; ++r) {
            mem3[r] = 0.9f * mem3[r] + cur3[r];
            ssum[r] += mem3[r];
        }
    }

    // ---------- write spk_sum / NUM_STEPS  ([B,4]) ----------
    if (nl < NC) {
        #pragma unroll
        for (int r = 0; r < 8; ++r)
            out[(size_t)(row0 + r + hi * 8) * NC + nl] = ssum[r] * (1.0f / STEPS);
    }
}

extern "C" void kernel_launch(void* const* d_in, const int* in_sizes, int n_in,
                              void* d_out, int out_size, void* d_ws, size_t ws_size,
                              hipStream_t stream) {
    const float* x  = (const float*)d_in[0];
    const float* w1 = (const float*)d_in[1];
    const float* b1 = (const float*)d_in[2];
    const float* w2 = (const float*)d_in[3];
    const float* b2 = (const float*)d_in[4];
    const float* w3 = (const float*)d_in[5];
    const float* b3 = (const float*)d_in[6];
    float* out = (float*)d_out;

    // Pass 1: pack weights into bf16 B-fragment layout (21 frags x 32 lanes).
    pack_weights<<<1, 672, 0, stream>>>(w1, w2, w3, (unsigned short*)d_ws);

    // Pass 2: fused SNN. 128 batch rows per 256-thread block.
    const int batch  = in_sizes[0] / IN_DIM;     // 131072
    const int blocks = batch / 128;
    snn_fused<<<blocks, 256, 0, stream>>>(x, b1, b2, b3,
                                          (const v16bf*)d_ws, out);
}
